// MultiHeadAttention_7121055776847
// MI455X (gfx1250) — matrix operand order repaired and checked by reading, not compiled, not run on hardware
//
#include <hip/hip_runtime.h>
#include <hip/hip_bf16.h>

// ---------------- types ----------------
typedef _Float16 v16h __attribute__((ext_vector_type(16)));
typedef _Float16 h8   __attribute__((ext_vector_type(8)));
typedef float    v8f  __attribute__((ext_vector_type(8)));

// ---------------- problem dims ----------------
#define BB   4
#define TT   2048
#define DD   768
#define HH   12
#define HD   64
#define BT   (BB*TT)          // 8192 rows
#define MT4  (BT/64)          // 128 row blocks (64 rows)
#define NT4  (DD/64)          // 12 col blocks (64 cols)

// workspace layout, in _Float16 elements (total ~67.6 MB)
#define OFF_X16   0u
#define OFF_WT    6291456u                 // 4 transposed weights, contiguous
#define OFF_Q     (OFF_WT + 4u*589824u)    // [B][H][T][HD]
#define OFF_K     (OFF_Q + 6291456u)       // [B][H][T][HD]
#define OFF_V     (OFF_K + 6291456u)       // [B][H][HD][T]  (transposed)
#define OFF_CTX   (OFF_V + 6291456u)       // [B][T][D]

// ---------------- helpers ----------------
__device__ __forceinline__ v16h pack16(h8 lo, h8 hi) {
  v16h r;
#pragma unroll
  for (int i = 0; i < 8; ++i) { r[i] = lo[i]; r[8 + i] = hi[i]; }
  return r;
}
__device__ __forceinline__ h8 ld8(const _Float16* p) { return *(const h8*)p; }

__device__ __forceinline__ v8f wmma_f16(v16h a, v16h b, v8f c) {
  return __builtin_amdgcn_wmma_f32_16x16x32_f16(false, a, false, b,
                                                (short)0, c, false, false);
}

// reductions across the 16 lanes that hold one C/D-matrix row (wave32)
__device__ __forceinline__ float rmax16(float v) {
  v = fmaxf(v, __shfl_xor(v, 1, 32));
  v = fmaxf(v, __shfl_xor(v, 2, 32));
  v = fmaxf(v, __shfl_xor(v, 4, 32));
  v = fmaxf(v, __shfl_xor(v, 8, 32));
  return v;
}
__device__ __forceinline__ float rsum16(float v) {
  v += __shfl_xor(v, 1, 32);
  v += __shfl_xor(v, 2, 32);
  v += __shfl_xor(v, 4, 32);
  v += __shfl_xor(v, 8, 32);
  return v;
}

// ---------------- stage 0: conversions ----------------
__global__ __launch_bounds__(256) void cvt_x_f16(const float* __restrict__ x,
                                                 _Float16* __restrict__ x16,
                                                 int n) {
  int i = blockIdx.x * 256 + threadIdx.x;
  if (i < n) x16[i] = (_Float16)x[i];
}

// transpose-convert all four weights: wT[w][n][k] = (f16) W_w[k][n]
__global__ __launch_bounds__(256) void cvt_wT_f16(const float* __restrict__ wq,
                                                  const float* __restrict__ wk,
                                                  const float* __restrict__ wv,
                                                  const float* __restrict__ wo,
                                                  _Float16* __restrict__ wT) {
  int i = blockIdx.x * 256 + threadIdx.x;   // 0 .. 4*589824-1
  int w = i / (DD * DD);
  int rem = i - w * (DD * DD);
  int n = rem / DD, k = rem - n * DD;
  const float* src = (w == 0) ? wq : (w == 1) ? wk : (w == 2) ? wv : wo;
  wT[i] = (_Float16)src[k * DD + n];
}

// ---------------- stage 1: fused QKV projection ----------------
// one wave per 64x64 output block (4x4 register-blocked WMMA tiles)
__global__ __launch_bounds__(256) void qkv_gemm(const _Float16* __restrict__ x16,
                                                const _Float16* __restrict__ wT,
                                                _Float16* __restrict__ q16,
                                                _Float16* __restrict__ k16,
                                                _Float16* __restrict__ vT16) {
  const int lane = threadIdx.x & 31;
  const int wave = blockIdx.x * 8 + (threadIdx.x >> 5);
  const int which = wave / (MT4 * NT4);
  int rem = wave - which * (MT4 * NT4);
  const int mt = rem / NT4, nt = rem - mt * NT4;
  const int hl = lane >> 4, l16 = lane & 15;

  const _Float16* arow[4];
  const _Float16* brow[4];
#pragma unroll
  for (int i = 0; i < 4; ++i) {
    arow[i] = x16 + (size_t)(mt * 64 + i * 16 + l16) * DD;
    brow[i] = wT + (size_t)which * (DD * DD) + (size_t)(nt * 64 + i * 16 + l16) * DD;
  }

  v8f acc[4][4];
#pragma unroll
  for (int i = 0; i < 4; ++i)
#pragma unroll
    for (int j = 0; j < 4; ++j) acc[i][j] = (v8f){};

  for (int kk = 0; kk < DD; kk += 32) {
    const int ka = kk + (hl ? 8 : 0);
    const int kb = kk + (hl ? 8 : 0);
    v16h a[4], b[4];
#pragma unroll
    for (int i = 0; i < 4; ++i) {
      a[i] = pack16(ld8(arow[i] + ka), ld8(arow[i] + ka + 16));
      b[i] = pack16(ld8(brow[i] + kb), ld8(brow[i] + kb + 16));
    }
#pragma unroll
    for (int i = 0; i < 4; ++i)
#pragma unroll
      for (int j = 0; j < 4; ++j)
        acc[i][j] = wmma_f16(a[i], b[j], acc[i][j]);
  }

#pragma unroll
  for (int i = 0; i < 4; ++i) {
#pragma unroll
    for (int r = 0; r < 8; ++r) {
      const int tg = mt * 64 + i * 16 + r + 8 * hl;   // global token
      const int bb = tg >> 11, tt = tg & (TT - 1);
#pragma unroll
      for (int j = 0; j < 4; ++j) {
        const int d = nt * 64 + j * 16 + l16;
        const int h = d >> 6, hd = d & 63;
        const _Float16 val = (_Float16)acc[i][j][r];
        const size_t bh = (size_t)bb * HH + h;
        if (which == 0)      q16[((bh * TT + tt) << 6) + hd] = val;
        else if (which == 1) k16[((bh * TT + tt) << 6) + hd] = val;
        else                 vT16[((bh << 6) + hd) * TT + tt] = val;
      }
    }
  }
}

// ---------------- stage 2: causal flash attention ----------------
// one wave per (b,h, 32-row q block); two 16-row tiles share K/V fragments
__device__ __forceinline__ void softmax_update(v8f& s0, v8f& s1,
                                               float* mrow, float* lrow,
                                               v8f* oacc,
                                               int rowg_base, int key0, int l16) {
#pragma unroll
  for (int r = 0; r < 8; ++r) {
    const int rowg = rowg_base + r;
    float v0 = s0[r] * 0.125f;                        // 1/sqrt(64)
    float v1 = s1[r] * 0.125f;
    if (key0 + l16 > rowg)      v0 = -1e30f;
    if (key0 + 16 + l16 > rowg) v1 = -1e30f;

    float tmax = rmax16(fmaxf(v0, v1));
    const float newm = fmaxf(mrow[r], tmax);
    const float alpha = __expf(mrow[r] - newm);
    const float p0 = __expf(v0 - newm);
    const float p1 = __expf(v1 - newm);
    lrow[r] = lrow[r] * alpha + rsum16(p0 + p1);
    mrow[r] = newm;
#pragma unroll
    for (int c = 0; c < 4; ++c) oacc[c][r] *= alpha;
    s0[r] = p0; s1[r] = p1;
  }
}

__global__ __launch_bounds__(256) void attn(const _Float16* __restrict__ q16,
                                            const _Float16* __restrict__ k16,
                                            const _Float16* __restrict__ vT16,
                                            _Float16* __restrict__ ctx16) {
  __shared__ __align__(16) _Float16 plds[8][2][16 * 32];  // P scratch per wave
  const int lane = threadIdx.x & 31;
  const int wib = threadIdx.x >> 5;
  const int wave = blockIdx.x * 8 + wib;
  const int qb = wave & 63;           // 32-row q block 0..63
  const int bh = wave >> 6;           // 0..47
  const int hl = lane >> 4, l16 = lane & 15;

  const _Float16* qbase = q16 + (size_t)bh * TT * HD;
  const _Float16* kbase = k16 + (size_t)bh * TT * HD;
  const _Float16* vbase = vT16 + (size_t)bh * HD * TT;
  _Float16* lpA = plds[wib][0];
  _Float16* lpB = plds[wib][1];

  // Q A-fragments for both 16-row tiles, hd 0..31 / 32..63 (loaded once)
  v16h aqA0, aqA1, aqB0, aqB1;
  {
    const int ka = hl ? 8 : 0;
    const _Float16* qrA = qbase + (size_t)(qb * 32 + l16) * HD;
    const _Float16* qrB = qbase + (size_t)(qb * 32 + 16 + l16) * HD;
    aqA0 = pack16(ld8(qrA + ka),      ld8(qrA + ka + 16));
    aqA1 = pack16(ld8(qrA + 32 + ka), ld8(qrA + 32 + ka + 16));
    aqB0 = pack16(ld8(qrB + ka),      ld8(qrB + ka + 16));
    aqB1 = pack16(ld8(qrB + 32 + ka), ld8(qrB + 32 + ka + 16));
  }

  float mrowA[8], lrowA[8], mrowB[8], lrowB[8];
  v8f oaccA[4], oaccB[4];
#pragma unroll
  for (int r = 0; r < 8; ++r) {
    mrowA[r] = mrowB[r] = -1e30f;
    lrowA[r] = lrowB[r] = 0.0f;
  }
#pragma unroll
  for (int c = 0; c < 4; ++c) { oaccA[c] = (v8f){}; oaccB[c] = (v8f){}; }

  const int rowgA = qb * 32 + 8 * hl;
  const int rowgB = qb * 32 + 16 + 8 * hl;
  const int nkb = qb + 1;             // exact causal bound for both tiles

  for (int kb = 0; kb < nkb; ++kb) {
    const int key0 = kb * 32;
    if (kb + 1 < nkb)
      __builtin_prefetch(kbase + (size_t)(key0 + 32 + l16) * HD, 0, 0);

    // ---- K fragments (shared by both q tiles) ----
    const int kofs = hl ? 8 : 0;
    const _Float16* kr0 = kbase + (size_t)(key0 + l16) * HD + kofs;
    const _Float16* kr1 = kbase + (size_t)(key0 + 16 + l16) * HD + kofs;
    v16h kf00 = pack16(ld8(kr0),      ld8(kr0 + 16));
    v16h kf01 = pack16(ld8(kr0 + 32), ld8(kr0 + 48));
    v16h kf10 = pack16(ld8(kr1),      ld8(kr1 + 16));
    v16h kf11 = pack16(ld8(kr1 + 32), ld8(kr1 + 48));

    // ---- S = Q * K^T ----
    v8f sA0 = {}, sA1 = {}, sB0 = {}, sB1 = {};
    sA0 = wmma_f16(aqA0, kf00, sA0); sA0 = wmma_f16(aqA1, kf01, sA0);
    sA1 = wmma_f16(aqA0, kf10, sA1); sA1 = wmma_f16(aqA1, kf11, sA1);
    sB0 = wmma_f16(aqB0, kf00, sB0); sB0 = wmma_f16(aqB1, kf01, sB0);
    sB1 = wmma_f16(aqB0, kf10, sB1); sB1 = wmma_f16(aqB1, kf11, sB1);

    // ---- online softmax per tile ----
    softmax_update(sA0, sA1, mrowA, lrowA, oaccA, rowgA, key0, l16);
    softmax_update(sB0, sB1, mrowB, lrowB, oaccB, rowgB, key0, l16);

    // ---- transpose P tiles (D-layout -> A-layout) through LDS ----
#pragma unroll
    for (int r = 0; r < 8; ++r) {
      const int m = r + 8 * hl;
      lpA[m * 32 + l16]      = (_Float16)sA0[r];
      lpA[m * 32 + 16 + l16] = (_Float16)sA1[r];
      lpB[m * 32 + l16]      = (_Float16)sB0[r];
      lpB[m * 32 + 16 + l16] = (_Float16)sB1[r];
    }
    asm volatile("s_wait_dscnt 0" ::: "memory");   // RAW through LDS
    const int pofs = l16 * 32 + (hl ? 8 : 0);
    v16h apA = pack16(ld8(lpA + pofs), ld8(lpA + pofs + 16));
    v16h apB = pack16(ld8(lpB + pofs), ld8(lpB + pofs + 16));

    // ---- O += P * V (V fragments shared by both q tiles) ----
#pragma unroll
    for (int c = 0; c < 4; ++c) {
      const _Float16* vr = vbase + (size_t)(c * 16 + l16) * TT + key0 + (hl ? 8 : 0);
      v16h bv = pack16(ld8(vr), ld8(vr + 16));
      oaccA[c] = wmma_f16(apA, bv, oaccA[c]);
      oaccB[c] = wmma_f16(apB, bv, oaccB[c]);
    }
  }

  // ---- normalize + store ctx as [B][T][D] f16 ----
  const int b = bh / HH, h = bh - b * HH;
#pragma unroll
  for (int r = 0; r < 8; ++r) {
    const float invA = 1.0f / lrowA[r];
    const float invB = 1.0f / lrowB[r];
    const int rgA = qb * 32 + r + 8 * hl;
    const int rgB = rgA + 16;
    const size_t baseA = ((size_t)b * TT + rgA) * DD + h * HD;
    const size_t baseB = ((size_t)b * TT + rgB) * DD + h * HD;
#pragma unroll
    for (int c = 0; c < 4; ++c) {
      ctx16[baseA + c * 16 + l16] = (_Float16)(oaccA[c][r] * invA);
      ctx16[baseB + c * 16 + l16] = (_Float16)(oaccB[c][r] * invB);
    }
  }
}

// ---------------- stage 3: output projection + bias ----------------
// one wave per 64x64 output block (4x4 register-blocked WMMA tiles)
__global__ __launch_bounds__(256) void oproj(const _Float16* __restrict__ ctx16,
                                             const _Float16* __restrict__ woT,
                                             const float* __restrict__ bo,
                                             float* __restrict__ out) {
  const int lane = threadIdx.x & 31;
  const int wave = blockIdx.x * 8 + (threadIdx.x >> 5);
  const int mt = wave / NT4, nt = wave - mt * NT4;
  const int hl = lane >> 4, l16 = lane & 15;

  const _Float16* arow[4];
  const _Float16* brow[4];
#pragma unroll
  for (int i = 0; i < 4; ++i) {
    arow[i] = ctx16 + (size_t)(mt * 64 + i * 16 + l16) * DD;
    brow[i] = woT + (size_t)(nt * 64 + i * 16 + l16) * DD;
  }

  v8f acc[4][4];
#pragma unroll
  for (int i = 0; i < 4; ++i)
#pragma unroll
    for (int j = 0; j < 4; ++j) acc[i][j] = (v8f){};

  for (int kk = 0; kk < DD; kk += 32) {
    const int ka = kk + (hl ? 8 : 0);
    const int kb = kk + (hl ? 8 : 0);
    v16h a[4], b[4];
#pragma unroll
    for (int i = 0; i < 4; ++i) {
      a[i] = pack16(ld8(arow[i] + ka), ld8(arow[i] + ka + 16));
      b[i] = pack16(ld8(brow[i] + kb), ld8(brow[i] + kb + 16));
    }
#pragma unroll
    for (int i = 0; i < 4; ++i)
#pragma unroll
      for (int j = 0; j < 4; ++j)
        acc[i][j] = wmma_f16(a[i], b[j], acc[i][j]);
  }

#pragma unroll
  for (int i = 0; i < 4; ++i) {
#pragma unroll
    for (int r = 0; r < 8; ++r) {
      const int m = mt * 64 + i * 16 + r + 8 * hl;
#pragma unroll
      for (int j = 0; j < 4; ++j) {
        const int n = nt * 64 + j * 16 + l16;
        out[(size_t)m * DD + n] = acc[i][j][r] + bo[n];
      }
    }
  }
}

// ---------------- host entry ----------------
extern "C" void kernel_launch(void* const* d_in, const int* in_sizes, int n_in,
                              void* d_out, int out_size, void* d_ws, size_t ws_size,
                              hipStream_t stream) {
  const float* x  = (const float*)d_in[0];
  const float* wq = (const float*)d_in[1];
  const float* wk = (const float*)d_in[2];
  const float* wv = (const float*)d_in[3];
  const float* wo = (const float*)d_in[4];
  const float* bo = (const float*)d_in[5];
  float* out = (float*)d_out;

  _Float16* ws = (_Float16*)d_ws;
  _Float16* x16  = ws + OFF_X16;
  _Float16* wT   = ws + OFF_WT;     // 4 contiguous transposed weights
  _Float16* q16  = ws + OFF_Q;
  _Float16* k16  = ws + OFF_K;
  _Float16* vT16 = ws + OFF_V;
  _Float16* ctx  = ws + OFF_CTX;

  cvt_x_f16<<<(BT * DD) / 256, 256, 0, stream>>>(x, x16, BT * DD);
  cvt_wT_f16<<<(4 * DD * DD) / 256, 256, 0, stream>>>(wq, wk, wv, wo, wT);
  qkv_gemm<<<(3 * MT4 * NT4) / 8, 256, 0, stream>>>(x16, wT, q16, k16, vT16);
  attn<<<(BB * HH * (TT / 32)) / 8, 256, 0, stream>>>(q16, k16, vT16, ctx);
  oproj<<<(MT4 * NT4) / 8, 256, 0, stream>>>(ctx, wT + 3u * DD * DD, bo, out);
}
